// StepwiseSAE_6622839570549
// MI455X (gfx1250) — matrix operand order repaired and checked by reading, not compiled, not run on hardware
//
#include <hip/hip_runtime.h>
#include <hip/hip_bf16.h>

typedef __attribute__((ext_vector_type(16))) __bf16 v16bf;
typedef __attribute__((ext_vector_type(8)))  float  v8f;

#define DM   768      // d_model
#define DS   24576    // d_sae
#define NT   4096     // n_tok
#define KTOP 64

// ---- Encode GEMM tiling ----
#define MT   64       // token rows per block
#define NTL  128      // feature cols per block
#define KT   32       // k per step (one bf16 WMMA depth)
#define LDSA 40       // bf16 elems per LDS row (32 + 8 pad -> 80B row stride)
#define KSTEPS (DM / KT)

__device__ __forceinline__ unsigned pack_bf16x2(float a, float b) {
  union { unsigned u; __bf16 h[2]; } p;
  p.h[0] = (__bf16)a;
  p.h[1] = (__bf16)b;
  return p.u;  // lowers to v_cvt_pk_bf16_f32 with both halves live
}

// =====================================================================
// Kernel 1: W_dec (768 x 24576) -> W_decT (24576 x 768), coalesced tiles
// =====================================================================
__global__ __launch_bounds__(256) void sae_transpose_wdec(
    const float* __restrict__ W_dec, float* __restrict__ wdT) {
  __shared__ float tile[32][33];
  const int sx = blockIdx.x * 32 + threadIdx.x;  // s index (contiguous read)
  const int d0 = blockIdx.y * 32 + threadIdx.y;  // d base
#pragma unroll
  for (int i = 0; i < 4; ++i)
    tile[threadIdx.y + 8 * i][threadIdx.x] = W_dec[(size_t)(d0 + 8 * i) * DS + sx];
  __syncthreads();
  const int dx = blockIdx.y * 32 + threadIdx.x;  // d (contiguous write)
  const int s0 = blockIdx.x * 32 + threadIdx.y;  // s base
#pragma unroll
  for (int i = 0; i < 4; ++i)
    wdT[(size_t)(s0 + 8 * i) * DM + dx] = tile[threadIdx.x][threadIdx.y + 8 * i];
}

// =====================================================================
// Kernel 2: pre = h @ W_enc^T + b_enc  via v_wmma_f32_16x16x32_bf16
//   NT-GEMM (both operands K-contiguous). 8 waves/block, 64x128 tile.
//   Double-buffered LDS with split staging:
//     global_load(k+1)->regs | ds_load frags(k) | wmma x4 | cvt+ds_store(k+1)
//   so WMMA's dscnt wait never covers stores that depend on in-flight HBM.
// =====================================================================
__global__ __launch_bounds__(256) void sae_encode(
    const float* __restrict__ h, const float* __restrict__ W_enc,
    const float* __restrict__ b_enc, float* __restrict__ pre) {
  __shared__ __attribute__((aligned(16))) __bf16 As[2 * MT * LDSA];
  __shared__ __attribute__((aligned(16))) __bf16 Bs[2 * NTL * LDSA];

  const int tid    = threadIdx.x;
  const int wave   = tid >> 5;
  const int lane   = tid & 31;
  const int blk_n  = blockIdx.x * NTL;
  const int blk_m  = blockIdx.y * MT;
  const int m_sub  = wave & 3;   // which 16-row strip of the 64-row tile
  const int n_half = wave >> 2;  // which 64-col half (4 subtiles each)

  union { v8f v; float f[8]; } acc[4];
#pragma unroll
  for (int j = 0; j < 4; ++j)
#pragma unroll
    for (int e = 0; e < 8; ++e) acc[j].f[e] = 0.0f;

  // per-thread staging coordinates (fixed across K-steps)
  const int ar0 = tid >> 3, ac4 = tid & 7;             // A: 2 float4 / thread
  const int br0 = tid >> 3;                            // B: 4 float4 / thread

  float4 ra[2], rb[4];

  auto stage_load = [&](int k0) {
#pragma unroll
    for (int i = 0; i < 2; ++i)
      ra[i] = *(const float4*)(h + (size_t)(blk_m + ar0 + 32 * i) * DM + k0 + 4 * ac4);
#pragma unroll
    for (int i = 0; i < 4; ++i)
      rb[i] = *(const float4*)(W_enc + (size_t)(blk_n + br0 + 32 * i) * DM + k0 + 4 * ac4);
  };

  auto stage_store = [&](int buf) {
    char* abase = (char*)As + (size_t)buf * MT * LDSA * 2;
    char* bbase = (char*)Bs + (size_t)buf * NTL * LDSA * 2;
#pragma unroll
    for (int i = 0; i < 2; ++i) {
      union { unsigned long long q; unsigned u[2]; } p;
      p.u[0] = pack_bf16x2(ra[i].x, ra[i].y);
      p.u[1] = pack_bf16x2(ra[i].z, ra[i].w);
      *(unsigned long long*)(abase + (ar0 + 32 * i) * (LDSA * 2) + 8 * ac4) = p.q;
    }
#pragma unroll
    for (int i = 0; i < 4; ++i) {
      union { unsigned long long q; unsigned u[2]; } p;
      p.u[0] = pack_bf16x2(rb[i].x, rb[i].y);
      p.u[1] = pack_bf16x2(rb[i].z, rb[i].w);
      *(unsigned long long*)(bbase + (br0 + 32 * i) * (LDSA * 2) + 8 * ac4) = p.q;
    }
  };

  stage_load(0);
  stage_store(0);
  __syncthreads();

  for (int kt = 0; kt < KSTEPS; ++kt) {
    const int cur = kt & 1;

    // (1) start next tile's HBM loads (results consumed only after the WMMAs)
    if (kt + 1 < KSTEPS) stage_load((kt + 1) * KT);
    if (kt + 2 < KSTEPS) {  // warm near caches two slabs ahead
      __builtin_prefetch(h + (size_t)(blk_m + (tid & 63)) * DM + (kt + 2) * KT, 0, 3);
      __builtin_prefetch(W_enc + (size_t)(blk_n + (tid & 127)) * DM + (kt + 2) * KT, 0, 3);
    }

    const char* abase = (const char*)As + (size_t)cur * MT * LDSA * 2;
    const char* bbase = (const char*)Bs + (size_t)cur * NTL * LDSA * 2;

    // (2) fragment ds_loads — the only DS ops outstanding at the WMMAs
    // A fragment: 16x32 bf16 ISA layout:
    //  lanes 0-15 (M=lane): K 0..7 in V0-3, K 16..23 in V4-7
    //  lanes 16-31 (M=lane-16): K 8..15 in V0-3, K 24..31 in V4-7
    union { v16bf v; unsigned u[8]; } afrag;
    {
      const int m    = lane & 15;
      const int koff = (lane >> 4) * 8;  // 0 or 8
      const unsigned* arow =
          (const unsigned*)(abase + (m_sub * 16 + m) * (LDSA * 2));
#pragma unroll
      for (int v = 0; v < 4; ++v) afrag.u[v]     = arow[(koff >> 1) + v];
#pragma unroll
      for (int v = 0; v < 4; ++v) afrag.u[4 + v] = arow[8 + (koff >> 1) + v];
    }
    // B fragments: 32x16 bf16: lanes 0-15 hold K 0..7,16..23 of col N=lane,
    // lanes 16-31 hold K 8..15,24..31 of col N=lane-16; 2 K-values per VGPR.
    union { v16bf v; unsigned u[8]; } bfrag[4];
    {
      const int n  = lane & 15;
      const int kb = (lane >> 4) * 8;  // 0 or 8
#pragma unroll
      for (int j = 0; j < 4; ++j) {
        const unsigned* brow =
            (const unsigned*)(bbase + (n_half * 64 + j * 16 + n) * (LDSA * 2));
#pragma unroll
        for (int v = 0; v < 8; ++v) bfrag[j].u[v] = brow[(kb >> 1) + (v & 3) + ((v >> 2) << 3)];
      }
    }

    // (3) matrix ops
#pragma unroll
    for (int j = 0; j < 4; ++j)
      acc[j].v = __builtin_amdgcn_wmma_f32_16x16x32_bf16(
          false, afrag.v, false, bfrag[j].v, (short)0, acc[j].v, false, false);

    // (4) convert + store next tile (waits loadcnt here, after compute)
    if (kt + 1 < KSTEPS) stage_store(cur ^ 1);

    __syncthreads();  // cur fully read + nxt fully written
  }

  // ---- epilogue: C layout lane l, VGPR v -> M = 8*(l>>4)+v, N = l&15 ----
  const int mrow = (lane >> 4) * 8;
  const int ncol = lane & 15;
#pragma unroll
  for (int j = 0; j < 4; ++j) {
    const int col = blk_n + n_half * 64 + j * 16 + ncol;
    const float be = b_enc[col];
#pragma unroll
    for (int v = 0; v < 8; ++v) {
      const int row = blk_m + m_sub * 16 + mrow + v;
      pre[(size_t)row * DS + col] = acc[j].f[v] + be;
    }
  }
}

// =====================================================================
// Kernel 3: exact per-row top-64 via 4-pass radix select on monotone keys.
// Row (96KB) cached in dynamic LDS (320KB/WGP). Rewrites z in place,
// emits compact (idx, relu(val)) list for the sparse decode.
// =====================================================================
__device__ __forceinline__ unsigned mono_key(unsigned b) {
  return (b & 0x80000000u) ? ~b : (b | 0x80000000u);
}

__global__ __launch_bounds__(256) void sae_topk(
    float* __restrict__ z, float* __restrict__ cvals, int* __restrict__ cidx) {
  extern __shared__ unsigned keys[];  // DS entries
  __shared__ unsigned hist[256];
  __shared__ unsigned s_prefix, s_mask, s_remaining, s_tie, s_slot;

  const int row = blockIdx.x;
  const int tid = threadIdx.x;
  float* zrow = z + (size_t)row * DS;

  for (int i = tid * 4; i < DS; i += 256 * 4) {
    float4 f = *(const float4*)(zrow + i);
    uint4 k;
    k.x = mono_key(__float_as_uint(f.x));
    k.y = mono_key(__float_as_uint(f.y));
    k.z = mono_key(__float_as_uint(f.z));
    k.w = mono_key(__float_as_uint(f.w));
    *(uint4*)(keys + i) = k;
  }
  if (tid == 0) { s_prefix = 0u; s_mask = 0u; s_remaining = KTOP; s_tie = 0u; s_slot = 0u; }
  __syncthreads();

  for (int shift = 24; shift >= 0; shift -= 8) {
    hist[tid] = 0u;
    __syncthreads();
    const unsigned prefix = s_prefix, mask = s_mask;
    for (int i = tid; i < DS; i += 256) {
      unsigned k = keys[i];
      if ((k & mask) == prefix) atomicAdd(&hist[(k >> shift) & 255u], 1u);
    }
    __syncthreads();
    if (tid == 0) {
      unsigned rem = s_remaining, cum = 0u;
      int b = 255;
      for (; b > 0; --b) {
        unsigned c = hist[b];
        if (cum + c >= rem) break;
        cum += c;
      }
      s_prefix    = prefix | ((unsigned)b << shift);
      s_mask      = mask | (0xFFu << shift);
      s_remaining = rem - cum;
    }
    __syncthreads();
  }

  const unsigned T = s_prefix;          // 64th-largest key
  const unsigned nties = s_remaining;   // how many == T to accept

  auto emit = [&](unsigned k, int i) -> float {
    bool sel = (k > T);
    if (!sel && k == T) sel = (atomicAdd(&s_tie, 1u) < nties);
    const unsigned bits = (k & 0x80000000u) ? (k & 0x7FFFFFFFu) : ~k;
    const float f = __uint_as_float(bits);
    if (sel) {
      unsigned p = atomicAdd(&s_slot, 1u);
      if (p < KTOP) {
        cvals[row * KTOP + p] = fmaxf(f, 0.0f);
        cidx[row * KTOP + p]  = i;
      }
      return fmaxf(f, 0.0f);
    }
    return 0.0f;
  };

  for (int i = tid * 4; i < DS; i += 256 * 4) {
    uint4 k = *(const uint4*)(keys + i);
    float4 o;
    o.x = emit(k.x, i + 0);
    o.y = emit(k.y, i + 1);
    o.z = emit(k.z, i + 2);
    o.w = emit(k.w, i + 3);
    *(float4*)(zrow + i) = o;
  }
}

// =====================================================================
// Kernel 4: sparse decode h_hat[n,:] = b_dec + sum_j val_j * W_decT[idx_j,:]
// 192 threads x float4: one coalesced b128 per row segment per term.
// W_decT (75MB) is L2-resident (192MB L2) across all 4096 tokens.
// =====================================================================
__global__ __launch_bounds__(192) void sae_decode(
    const float* __restrict__ wdT, const float* __restrict__ cvals,
    const int* __restrict__ cidx, const float* __restrict__ b_dec,
    float* __restrict__ hhat) {
  __shared__ float sval[KTOP];
  __shared__ int   sidx[KTOP];
  const int row = blockIdx.x;
  const int tid = threadIdx.x;
  if (tid < KTOP) {
    sval[tid] = cvals[row * KTOP + tid];
    sidx[tid] = cidx[row * KTOP + tid];
  }
  __syncthreads();
  const int d4 = tid * 4;                       // 192 * 4 = 768 = DM
  float4 a = *(const float4*)(b_dec + d4);
#pragma unroll 4
  for (int j = 0; j < KTOP; ++j) {
    const float w = sval[j];
    const float4 r = *(const float4*)(wdT + (size_t)sidx[j] * DM + d4);
    a.x = fmaf(w, r.x, a.x);
    a.y = fmaf(w, r.y, a.y);
    a.z = fmaf(w, r.z, a.z);
    a.w = fmaf(w, r.w, a.w);
  }
  *(float4*)(hhat + (size_t)row * DM + d4) = a;
}

extern "C" void kernel_launch(void* const* d_in, const int* in_sizes, int n_in,
                              void* d_out, int out_size, void* d_ws, size_t ws_size,
                              hipStream_t stream) {
  const float* h     = (const float*)d_in[0];
  const float* W_enc = (const float*)d_in[1];
  const float* b_enc = (const float*)d_in[2];
  const float* W_dec = (const float*)d_in[3];
  const float* b_dec = (const float*)d_in[4];

  float* out  = (float*)d_out;
  float* hhat = out;                       // (NT, DM)
  float* z    = out + (size_t)NT * DM;     // (NT, DS); pre lives here first

  float* wdT   = (float*)d_ws;                        // DS*DM floats (75.5 MB)
  float* cvals = wdT + (size_t)DS * DM;               // NT*KTOP floats
  int*   cidx  = (int*)(cvals + (size_t)NT * KTOP);   // NT*KTOP ints

  (void)hipFuncSetAttribute((const void*)sae_topk,
                            hipFuncAttributeMaxDynamicSharedMemorySize,
                            (int)(DS * sizeof(unsigned)));

  sae_transpose_wdec<<<dim3(DS / 32, DM / 32), dim3(32, 8), 0, stream>>>(W_dec, wdT);
  sae_encode<<<dim3(DS / NTL, NT / MT), 256, 0, stream>>>(h, W_enc, b_enc, z);
  sae_topk<<<NT, 256, DS * sizeof(unsigned), stream>>>(z, cvals, cidx);
  sae_decode<<<NT, 192, 0, stream>>>(wdT, cvals, cidx, b_dec, hhat);
}
